// LTFGW_34892314312764
// MI455X (gfx1250) — matrix operand order repaired and checked by reading, not compiled, not run on hardware
//
#include <hip/hip_runtime.h>
#include <hip/hip_bf16.h>
#include <stdint.h>

typedef __attribute__((ext_vector_type(16))) _Float16 v16h;
typedef __attribute__((ext_vector_type(8)))  _Float16 v8h;
typedef __attribute__((ext_vector_type(8)))  float    v8f;

#define KNBR 16
#define K1   17
#define NTPL 10
#define TN   10
#define NCOL 100    // NTPL * TN flattened template-node columns
#define MSTR 113    // odd LDS row stride (floats) -> conflict-free column walks
#define DIMS 128

// ---------------- preprocessing kernels ----------------

__global__ void k_init(int* cnt, int* cursor, int n) {
  int i = blockIdx.x * blockDim.x + threadIdx.x;
  if (i < n) { cnt[i] = 0; cursor[i] = 0; }
}

__global__ void k_count(const int* __restrict__ src, int* cnt, int E, int n) {
  int e = blockIdx.x * blockDim.x + threadIdx.x;
  if (e < E) {
    int s = src[e];
    if (s >= 0 && s < n) atomicAdd(&cnt[s], 1);
  }
}

__global__ void k_scan(const int* __restrict__ cnt, int* __restrict__ start, int n) {
  __shared__ int part[1024];
  int t = threadIdx.x;
  int chunk = (n + 1023) >> 10;
  int base = t * chunk;
  int s = 0;
  for (int j = 0; j < chunk; ++j) { int idx = base + j; if (idx < n) s += cnt[idx]; }
  part[t] = s;
  __syncthreads();
  for (int off = 1; off < 1024; off <<= 1) {
    int v = (t >= off) ? part[t - off] : 0;
    __syncthreads();
    part[t] += v;
    __syncthreads();
  }
  int run = (t == 0) ? 0 : part[t - 1];
  for (int j = 0; j < chunk; ++j) {
    int idx = base + j;
    if (idx < n) { start[idx] = run; run += cnt[idx]; }
  }
  if (t == 1023) start[n] = part[1023];
}

__global__ void k_scatter(const int* __restrict__ src, const int* __restrict__ start,
                          int* cursor, int* __restrict__ bucket, int E, int n) {
  int e = blockIdx.x * blockDim.x + threadIdx.x;
  if (e < E) {
    int s = src[e];
    if (s >= 0 && s < n) {
      int pos = start[s] + atomicAdd(&cursor[s], 1);
      bucket[pos] = e;
    }
  }
}

// keep the KNBR smallest edge-ids per node (== first K of a stable sort by src)
__global__ void k_select(const int* __restrict__ dst, const int* __restrict__ cnt,
                         const int* __restrict__ start, const int* __restrict__ bucket,
                         int* __restrict__ nbrs, int* __restrict__ nvalid, int n) {
  int i = blockIdx.x * blockDim.x + threadIdx.x;
  if (i >= n) return;
  int s0 = start[i], c = cnt[i];
  int keep = c < KNBR ? c : KNBR;
  int ids[KNBR];
  int m = 0;
  for (int j = 0; j < c; ++j) {
    int e = bucket[s0 + j];
    if (m < KNBR) {
      int p = m++;
      while (p > 0 && ids[p - 1] > e) { ids[p] = ids[p - 1]; --p; }
      ids[p] = e;
    } else if (e < ids[KNBR - 1]) {
      int p = KNBR - 1;
      while (p > 0 && ids[p - 1] > e) { ids[p] = ids[p - 1]; --p; }
      ids[p] = e;
    }
  }
  for (int j = 0; j < KNBR; ++j)
    nbrs[i * KNBR + j] = (j < keep) ? dst[ids[j]] : 0;
  nvalid[i] = keep;
}

// adjacency bitmask of the 17-node local subgraph: bit b of row a set iff
// kept-edge S[a] -> S[b] exists (rows/cols masked by validity)
__global__ void k_adj(const int* __restrict__ nbrs, const int* __restrict__ nvalid,
                      unsigned* __restrict__ adj, int n) {
  int tid = blockIdx.x * blockDim.x + threadIdx.x;
  if (tid >= n * K1) return;
  int i = tid / K1, a = tid % K1;
  int nv = nvalid[i];
  int cnt1 = nv + 1;
  if (a >= cnt1) { adj[tid] = 0u; return; }
  int S[K1];
  S[0] = i;
  for (int j = 0; j < KNBR; ++j) S[j + 1] = (j < nv) ? nbrs[i * KNBR + j] : -1;
  int u = S[a];
  int nu = nvalid[u];
  unsigned bits = 0u;
  for (int j = 0; j < nu; ++j) {
    int w = nbrs[u * KNBR + j];
    for (int b = 0; b < cnt1; ++b)
      if (S[b] == w) bits |= (1u << b);
  }
  adj[tid] = bits;
}

// template-side precompute: fp16 copy of F2 (padded to 112 rows), |F2|^2, hC2
__global__ void k_prep(const float* __restrict__ tmpl, const float* __restrict__ tf,
                       _Float16* __restrict__ f2h, float* __restrict__ n2g,
                       float* __restrict__ hc2) {
  int t = threadIdx.x;
  for (int idx = t; idx < 112 * DIMS; idx += blockDim.x) {
    int r = idx >> 7;
    float v = (r < NCOL) ? tf[idx] : 0.f;
    f2h[idx] = (_Float16)v;
  }
  for (int j = t; j < NCOL; j += blockDim.x) {
    const float* row = tf + (long)j * DIMS;
    float s = 0.f;
    for (int d = 0; d < DIMS; ++d) s += row[d] * row[d];
    n2g[j] = s;
    const float* c2r = tmpl + (long)j * TN;   // j = T*10+b -> C2[T][b][:]
    float h = 0.f;
    for (int b = 0; b < TN; ++b) h += c2r[b] * c2r[b];
    hc2[j] = 0.1f * h;                        // * q
  }
}

// ---------------- main fused FGW kernel: 1 block (10 waves) per node --------

__global__ __launch_bounds__(320)
void k_main(const float* __restrict__ x, const float* __restrict__ tmpl,
            const int* __restrict__ nbrs, const int* __restrict__ nvalid,
            const unsigned* __restrict__ adjg, const float* __restrict__ n2g,
            const float* __restrict__ hc2g, const _Float16* __restrict__ f2h,
            float* __restrict__ out, int n) {
  __shared__ __align__(16) _Float16 sA[32][136];   // padded F1 (fp16)
  __shared__ float sM[32][MSTR];                    // feature cost (full padded tile)
  __shared__ float sG[K1][MSTR];                    // stores logK = -G/reg
  __shared__ float sT[K1][MSTR];                    // transport plan
  __shared__ float sTmp[NTPL][K1][12];              // C1 @ T per template
  __shared__ float sC2[NTPL][TN][TN];
  __shared__ float sHC2[112];
  __shared__ float sN2[112];
  __shared__ float sN1[32];
  __shared__ float sCC1[K1];
  __shared__ int   sS[K1];
  __shared__ unsigned sAdj[K1];

  const int i = blockIdx.x;
  const int t = threadIdx.x;
  const int lane = t & 31;
  const int wv = t >> 5;
  const int nv = nvalid[i];
  const int cnt1 = nv + 1;

  if (t < K1) {
    int u = (t == 0) ? i : ((t - 1) < nv ? nbrs[i * KNBR + (t - 1)] : 0);
    sS[t] = u;
    unsigned ab = adjg[i * K1 + t];
    sAdj[t] = ab;
    sCC1[t] = (float)__popc(ab) / (float)cnt1;      // sum_b C1^2[a,b] * p[b]
  }
  for (int j = t; j < 112; j += 320) {              // zero-filled tails -> padding is finite
    sN2[j]  = (j < NCOL) ? n2g[j]  : 0.f;
    sHC2[j] = (j < NCOL) ? hc2g[j] : 0.f;
  }
  for (int j = t; j < NTPL * TN * TN; j += 320) ((float*)sC2)[j] = tmpl[j];

  if (t < 32) {                                     // fill F1 rows (fp16) + |F1|^2
    float nn = 0.f;
    if (t < cnt1) {
      const float* xr = x + (long)sS[t] * DIMS;
      for (int d = 0; d < DIMS; ++d) { float v = xr[d]; sA[t][d] = (_Float16)v; nn += v * v; }
    } else {
      for (int d = 0; d < DIMS; ++d) sA[t][d] = (_Float16)0.f;
    }
    sN1[t] = nn;
  }
  const float pself = 1.f / (float)cnt1;
  for (int j = t; j < K1 * NCOL; j += 320) {        // T0 = p x q
    int a = j / NCOL, c = j % NCOL;
    sT[a][c] = (a < cnt1) ? pself * 0.1f : 0.f;
  }
  __syncthreads();

  // ---- WMMA phase: M[a][col] = |F1a|^2 + |F2c|^2 - 2 F1a.F2c ----
  // 2 M-tiles x 7 N-tiles over 10 waves, K = 128 in 4 steps of 32 (f16, f32 acc).
  // Write-back is unconditional into the padded 32x112 footprint (no exec-mask churn).
  for (int tid = wv; tid < 14; tid += 10) {
    int mt = tid / 7, nt = tid % 7;
    int r   = mt * 16 + (lane & 15);
    int hiA = (lane & 16) ? 8 : 0;
    int bc  = nt * 16 + (lane & 15);
    int hiB = (lane & 16) ? 8 : 0;
    v8f acc = {0.f, 0.f, 0.f, 0.f, 0.f, 0.f, 0.f, 0.f};
#pragma unroll
    for (int kt = 0; kt < 4; ++kt) {
      int k0 = kt * 32;
      v8h a0 = *(const v8h*)&sA[r][k0 + hiA];
      v8h a1 = *(const v8h*)&sA[r][k0 + 16 + hiA];
      v8h b0 = *(const v8h*)&f2h[(long)bc * DIMS + k0 + hiB];
      v8h b1 = *(const v8h*)&f2h[(long)bc * DIMS + k0 + 16 + hiB];
      v16h av, bv;
#pragma unroll
      for (int q = 0; q < 8; ++q) { av[q] = a0[q]; av[q + 8] = a1[q]; bv[q] = b0[q]; bv[q + 8] = b1[q]; }
      acc = __builtin_amdgcn_wmma_f32_16x16x32_f16(false, av, false, bv, (short)0, acc, false, false);
    }
    int row0 = mt * 16 + ((lane & 16) ? 8 : 0);
    int col  = nt * 16 + (lane & 15);
#pragma unroll
    for (int v = 0; v < 8; ++v)
      sM[row0 + v][col] = sN1[row0 + v] + sN2[col] - 2.f * acc[v];
  }
  __syncthreads();

  // ---- FW + log-domain Sinkhorn: wave w owns template w (cols c0..c0+9) ----
  const int c0 = wv * TN;
  const float logq = -2.302585093f;                 // -ln 10
  const float lcnt = __logf((float)cnt1);
  const int bb = lane < TN ? lane : TN - 1;
  const int aa = lane < K1 ? lane : K1 - 1;
  const float logp = (lane < cnt1) ? -lcnt : -1e9f;
  float f = 0.f, g = 0.f;

  for (int k = 0; k < 5; ++k) {
    // tmp = C1 @ T  (bitmask-driven)
    for (int idx = lane; idx < K1 * TN; idx += 32) {
      int a = idx / TN, d = idx % TN;
      unsigned mb = sAdj[a];
      float s = 0.f;
      while (mb) { int c = __ffs(mb) - 1; mb &= mb - 1; s += sT[c][c0 + d]; }
      sTmp[wv][a][d] = s;
    }
    __syncthreads();
    // logK = -10 * (0.5*M + constC - 2*(tmp @ C2^T))
    for (int idx = lane; idx < K1 * TN; idx += 32) {
      int a = idx / TN, b = idx % TN;
      float cr = 0.f;
#pragma unroll
      for (int d = 0; d < TN; ++d) cr += sTmp[wv][a][d] * sC2[wv][b][d];
      float tens = sCC1[a] + sHC2[c0 + b] - 2.f * cr;
      sG[a][c0 + b] = -10.f * (0.5f * sM[a][c0 + b] + tens);
    }
    __syncthreads();
    // Sinkhorn (20 iters), f/g in registers, cross-lane via shfl, no barriers
    f = 0.f; g = 0.f;
    for (int it = 0; it < 20; ++it) {
      float m = -3.0e38f, s = 0.f;
      for (int a2 = 0; a2 < K1; ++a2) {             // g-step (lanes < 10 keep)
        float fa = __shfl(f, a2);
        float v = sG[a2][c0 + bb] + fa;
        float mn = fmaxf(m, v);
        s = s * __expf(m - mn) + __expf(v - mn);
        m = mn;
      }
      float gn = logq - (m + __logf(s));
      if (lane < TN) g = gn;
      m = -3.0e38f; s = 0.f;
      for (int b2 = 0; b2 < TN; ++b2) {             // f-step (lanes < 17 keep)
        float gb = __shfl(g, b2);
        float v = sG[aa][c0 + b2] + gb;
        float mn = fmaxf(m, v);
        s = s * __expf(m - mn) + __expf(v - mn);
        m = mn;
      }
      float fn = logp - (m + __logf(s));
      if (lane < K1) f = fn;
    }
    // T = (1-gamma)*T + gamma * exp(f + g + logK); keep loop wave-uniform for shfl
    float gamma = 2.f / ((float)k + 2.f);
    for (int idx = lane; idx < 192; idx += 32) {
      int a = idx / TN; if (a > K1 - 1) a = K1 - 1;
      int b = idx % TN;
      float fa = __shfl(f, a);
      float gb = __shfl(g, b);
      if (idx < K1 * TN) {
        float tn = __expf(fa + gb + sG[a][c0 + b]);
        sT[a][c0 + b] = (1.f - gamma) * sT[a][c0 + b] + gamma * tn;
      }
    }
    __syncthreads();
  }

  // ---- final: dist = 0.5*sum((M + tens) * T) ----
  for (int idx = lane; idx < K1 * TN; idx += 32) {
    int a = idx / TN, d = idx % TN;
    unsigned mb = sAdj[a];
    float s = 0.f;
    while (mb) { int c = __ffs(mb) - 1; mb &= mb - 1; s += sT[c][c0 + d]; }
    sTmp[wv][a][d] = s;
  }
  __syncthreads();
  float part = 0.f;
  for (int idx = lane; idx < K1 * TN; idx += 32) {
    int a = idx / TN, b = idx % TN;
    float cr = 0.f;
#pragma unroll
    for (int d = 0; d < TN; ++d) cr += sTmp[wv][a][d] * sC2[wv][b][d];
    float tens = sCC1[a] + sHC2[c0 + b] - 2.f * cr;
    part += sT[a][c0 + b] * 0.5f * (sM[a][c0 + b] + tens);
  }
#pragma unroll
  for (int off = 16; off; off >>= 1) part += __shfl_down(part, off);
  if (lane == 0) out[(long)i * NTPL + wv] = part;
}

// ---------------- host launcher ----------------

extern "C" void kernel_launch(void* const* d_in, const int* in_sizes, int n_in,
                              void* d_out, int out_size, void* d_ws, size_t ws_size,
                              hipStream_t stream) {
  const float* x    = (const float*)d_in[0];
  const int*   ei   = (const int*)d_in[1];
  const float* tmpl = (const float*)d_in[2];
  const float* tf   = (const float*)d_in[3];
  float* out = (float*)d_out;

  const int n = in_sizes[0] / DIMS;
  const int E = in_sizes[1] / 2;
  const int* src = ei;
  const int* dst = ei + E;

  // workspace carve-out
  int* cnt      = (int*)d_ws;              // n
  int* cursor   = cnt + n;                 // n
  int* start    = cursor + n;              // n+1
  int* nvalid   = start + (n + 1);         // n
  int* nbrs     = nvalid + n;              // n*KNBR
  unsigned* adj = (unsigned*)(nbrs + (long)n * KNBR);  // n*K1
  int* bucket   = (int*)(adj + (long)n * K1);          // E
  float* n2g    = (float*)(bucket + E);    // 112
  float* hc2    = n2g + 112;               // 112
  _Float16* f2h = (_Float16*)(((uintptr_t)(hc2 + 112) + 15) & ~(uintptr_t)15);

  k_init<<<(n + 255) / 256, 256, 0, stream>>>(cnt, cursor, n);
  k_count<<<(E + 255) / 256, 256, 0, stream>>>(src, cnt, E, n);
  k_scan<<<1, 1024, 0, stream>>>(cnt, start, n);
  k_scatter<<<(E + 255) / 256, 256, 0, stream>>>(src, start, cursor, bucket, E, n);
  k_select<<<(n + 255) / 256, 256, 0, stream>>>(dst, cnt, start, bucket, nbrs, nvalid, n);
  k_adj<<<(n * K1 + 255) / 256, 256, 0, stream>>>(nbrs, nvalid, adj, n);
  k_prep<<<1, 256, 0, stream>>>(tmpl, tf, f2h, n2g, hc2);
  k_main<<<n, 320, 0, stream>>>(x, tmpl, nbrs, nvalid, adj, n2g, hc2, f2h, out, n);
  (void)n_in; (void)out_size; (void)ws_size;
}
